// Attention_90898687852593
// MI455X (gfx1250) — matrix operand order repaired and checked by reading, not compiled, not run on hardware
//
#include <hip/hip_runtime.h>

// ---------------------------------------------------------------------------
// Problem constants (from reference): B=2048, D=1024, H=8, DH=64, INNER=512
// N = B*H = 16384 attention rows, head dim 64.
// ---------------------------------------------------------------------------
#define BATCH   2048
#define DMODEL  1024
#define INNER   512
#define NQ      16384   // B*H
#define DH      64
#define CHUNK   64      // keys per flash-attention iteration

typedef __attribute__((ext_vector_type(16))) __bf16 v16bf;
typedef __attribute__((ext_vector_type(8)))  __bf16 v8bf;
typedef __attribute__((ext_vector_type(8)))  float  v8f;
typedef __attribute__((ext_vector_type(4)))  int    v4i;

union V16 { v16bf v; v8bf h[2]; };

// --- CDNA5 async global->LDS path (ASYNCcnt), with safe fallback ------------
#if defined(__has_builtin)
#if __has_builtin(__builtin_amdgcn_global_load_async_to_lds_b128) && \
    __has_builtin(__builtin_amdgcn_s_wait_asynccnt)
#define HAVE_ASYNC_LDS 1
#endif
#endif
#ifndef HAVE_ASYNC_LDS
#define HAVE_ASYNC_LDS 0
#endif

static __device__ __forceinline__ void async16(const __bf16* g, __bf16* l) {
#if HAVE_ASYNC_LDS
  // Param 0 is v4i in AS(1) (prints as "int __vector(4) __device__ *"),
  // param 1 is the LDS destination in AS(3).
  typedef __attribute__((address_space(1))) v4i as1_v4i;
  typedef __attribute__((address_space(3))) v4i as3_v4i;
  __builtin_amdgcn_global_load_async_to_lds_b128((as1_v4i*)g, (as3_v4i*)l, 0, 0);
#else
  *(v8bf*)l = *(const v8bf*)g;
#endif
}

static __device__ __forceinline__ void wait_async() {
#if HAVE_ASYNC_LDS
  __builtin_amdgcn_s_wait_asynccnt(0);
#endif
}

static __device__ __forceinline__ v8f wmma_bf16(v16bf a, v16bf b, v8f c) {
  // (neg_a, A, neg_b, B, c_mod, C, reuse_a, reuse_b)
  return __builtin_amdgcn_wmma_f32_16x16x32_bf16(false, a, false, b, (short)0, c,
                                                 false, false);
}

static __device__ __forceinline__ float rowmax16(float v) {
#pragma unroll
  for (int i = 1; i < 16; i <<= 1) v = fmaxf(v, __shfl_xor(v, i, 32));
  return v;
}
static __device__ __forceinline__ float rowsum16(float v) {
#pragma unroll
  for (int i = 1; i < 16; i <<= 1) v += __shfl_xor(v, i, 32);
  return v;
}

// ---------------------------------------------------------------------------
// 1) fp32 -> bf16 convert (grid-stride)
// ---------------------------------------------------------------------------
__global__ __launch_bounds__(256) void k_cvt(const float* __restrict__ src,
                                             __bf16* __restrict__ dst, int n) {
  for (int i = blockIdx.x * blockDim.x + threadIdx.x; i < n;
       i += gridDim.x * blockDim.x)
    dst[i] = (__bf16)src[i];
}

// fp32 RxC -> bf16 transposed CxR (dst row = original column)
__global__ __launch_bounds__(256) void k_cvt_tr(const float* __restrict__ src,
                                                __bf16* __restrict__ dst,
                                                int R, int C) {
  int n = R * C;
  for (int i = blockIdx.x * blockDim.x + threadIdx.x; i < n;
       i += gridDim.x * blockDim.x) {
    int r = i / C, c = i - r * C;
    dst[(size_t)c * R + r] = (__bf16)src[i];
  }
}

// ---------------------------------------------------------------------------
// 2) bf16 WMMA GEMM: out = scale * (A[MxK] @ BT^T), BT is [N x K] (row = out col)
//    Each wave computes a 16x64 strip (4 WMMA N-tiles, K-loop step 32).
//    vmode 0: out bf16 row-major (ldc); vmode 1: scatter for V-transpose:
//             out[(col&63)*NQ + row*8 + (col>>6)]   (Vt is [DH x NQ])
// ---------------------------------------------------------------------------
__global__ __launch_bounds__(128) void k_gemm_bf16(
    const __bf16* __restrict__ A, const __bf16* __restrict__ BT,
    __bf16* __restrict__ out, int M, int N, int K, int ldc, float scale,
    int vmode) {
  const int lane = threadIdx.x & 31;
  const int wv   = threadIdx.x >> 5;
  const int w    = blockIdx.x * 4 + wv;
  const int strips_n = N >> 6;
  const int mt = w / strips_n;
  const int nt = w - mt * strips_n;
  if (mt >= (M >> 4)) return;

  const int half = lane >> 4, l15 = lane & 15;
  const int kofs = half * 8;                 // A-operand sub-lane K offset
  const size_t arow = (size_t)(mt * 16 + l15) * K;

  v8f c[4] = {v8f{}, v8f{}, v8f{}, v8f{}};

  for (int kk = 0; kk < K; kk += 32) {
    V16 a;
    a.h[0] = *(const v8bf*)(A + arow + kk + kofs);
    a.h[1] = *(const v8bf*)(A + arow + kk + 16 + kofs);
    const int koff = kk + half * 8;          // B-operand: lane<16 K0..7,16..23, else K8..15,24..31
#pragma unroll
    for (int t = 0; t < 4; ++t) {
      const __bf16* bp = BT + (size_t)(nt * 64 + t * 16 + l15) * K + koff;
      V16 b;
      b.h[0] = *(const v8bf*)bp;
      b.h[1] = *(const v8bf*)(bp + 16);
      c[t] = wmma_bf16(a.v, b.v, c[t]);
    }
  }

#pragma unroll
  for (int t = 0; t < 4; ++t) {
    const int col = nt * 64 + t * 16 + l15;
#pragma unroll
    for (int r = 0; r < 8; ++r) {
      const int row = mt * 16 + half * 8 + r;
      const float v = c[t][r] * scale;
      if (vmode == 0)
        out[(size_t)row * ldc + col] = (__bf16)v;
      else
        out[(size_t)(col & 63) * NQ + row * 8 + (col >> 6)] = (__bf16)v;
    }
  }
}

// ---------------------------------------------------------------------------
// 3) Flash attention: one wave = 16 query rows, loop over 256 chunks of 64 keys.
//    Double-buffered async global->LDS staging (ASYNCcnt), one barrier +
//    one s_wait_asynccnt per chunk. Q carries the 1/sqrt(DH) scale already.
//    P round-trips through per-wave LDS to convert C/D layout -> A layout.
// ---------------------------------------------------------------------------
__global__ __launch_bounds__(128) void k_flash(
    const __bf16* __restrict__ Qb, const __bf16* __restrict__ Kb,
    const __bf16* __restrict__ Vt, __bf16* __restrict__ Ob) {
  __shared__ __bf16 ldsK[2][CHUNK * 64];   // [buf][key][d]      2 x 8 KB
  __shared__ __bf16 ldsV[2][64 * CHUNK];   // [buf][d][key]      2 x 8 KB
  __shared__ __bf16 ldsP[4][16 * 72];      // per-wave P tile, 144B row stride

  const int lane = threadIdx.x & 31;
  const int wv   = threadIdx.x >> 5;
  const int half = lane >> 4, l15 = lane & 15;
  const int kofs = half * 8;
  const int qbase = (blockIdx.x * 4 + wv) * 16;

  // Q A-operands for contraction dims 0..31 / 32..63 (resident in VGPRs)
  V16 aq0, aq1;
  {
    const __bf16* qr = Qb + (size_t)(qbase + l15) * DH;
    aq0.h[0] = *(const v8bf*)(qr + kofs);
    aq0.h[1] = *(const v8bf*)(qr + 16 + kofs);
    aq1.h[0] = *(const v8bf*)(qr + 32 + kofs);
    aq1.h[1] = *(const v8bf*)(qr + 48 + kofs);
  }

  float m[8], l[8];
#pragma unroll
  for (int r = 0; r < 8; ++r) { m[r] = -3.0e38f; l[r] = 0.0f; }
  v8f o[4] = {v8f{}, v8f{}, v8f{}, v8f{}};

  // Cooperative stage of one 64-key chunk: K (64x64) and Vt (64x64), 16 KB.
  // 128 threads x 128 B each (8 x b128 async loads).
  auto stage = [&](int buf, int kc) {
    const int t   = threadIdx.x;
    const int row = t >> 1;            // 0..63
    const int cb  = (t & 1) * 32;      // element offset within 64-wide row
    const __bf16* gk = Kb + (size_t)(kc + row) * 64 + cb;
    const __bf16* gv = Vt + (size_t)row * NQ + kc + cb;
    __bf16* lk = &ldsK[buf][row * 64 + cb];
    __bf16* lv = &ldsV[buf][row * 64 + cb];
#pragma unroll
    for (int j = 0; j < 4; ++j) {
      async16(gk + j * 8, lk + j * 8);
      async16(gv + j * 8, lv + j * 8);
    }
  };

  stage(0, 0);

  for (int it = 0; it < NQ / CHUNK; ++it) {
    const int buf = it & 1;
    wait_async();          // this wave's async writes into ldsK/ldsV[buf] done
    __syncthreads();       // everyone's writes visible; prev buffer free

    if (it + 1 < NQ / CHUNK) stage(buf ^ 1, (it + 1) * CHUNK);

    const __bf16* Kc = ldsK[buf];
    const __bf16* Vc = ldsV[buf];

    // --- S = Q @ K^T : 4 key-tiles x (K=64 in 2 WMMA steps) ---
    v8f s[4] = {v8f{}, v8f{}, v8f{}, v8f{}};
#pragma unroll
    for (int t2 = 0; t2 < 4; ++t2) {
#pragma unroll
      for (int ks = 0; ks < 2; ++ks) {
        const __bf16* bp = Kc + (t2 * 16 + l15) * 64 + ks * 32 + half * 8;
        V16 b;
        b.h[0] = *(const v8bf*)bp;
        b.h[1] = *(const v8bf*)(bp + 16);
        s[t2] = wmma_bf16(ks ? aq1.v : aq0.v, b.v, s[t2]);
      }
    }

    // --- online softmax (row r lives in VGPR r across a 16-lane half) ---
    __bf16* Pw = ldsP[wv];
#pragma unroll
    for (int r = 0; r < 8; ++r) {
      float mx = fmaxf(fmaxf(s[0][r], s[1][r]), fmaxf(s[2][r], s[3][r]));
      mx = rowmax16(mx);
      const float mn = fmaxf(m[r], mx);
      const float alpha = __expf(m[r] - mn);
      const float p0 = __expf(s[0][r] - mn);
      const float p1 = __expf(s[1][r] - mn);
      const float p2 = __expf(s[2][r] - mn);
      const float p3 = __expf(s[3][r] - mn);
      l[r] = l[r] * alpha + rowsum16(p0 + p1 + p2 + p3);
      m[r] = mn;
      o[0][r] *= alpha; o[1][r] *= alpha; o[2][r] *= alpha; o[3][r] *= alpha;
      __bf16* pr = Pw + (half * 8 + r) * 72 + l15;
      pr[0]  = (__bf16)p0;
      pr[16] = (__bf16)p1;
      pr[32] = (__bf16)p2;
      pr[48] = (__bf16)p3;
    }

    // --- reload P (16x64) in A-operand layout: 2 operands of K=32 ---
    V16 ap0, ap1;
    ap0.h[0] = *(const v8bf*)(Pw + l15 * 72 + kofs);
    ap0.h[1] = *(const v8bf*)(Pw + l15 * 72 + 16 + kofs);
    ap1.h[0] = *(const v8bf*)(Pw + l15 * 72 + 32 + kofs);
    ap1.h[1] = *(const v8bf*)(Pw + l15 * 72 + 48 + kofs);

    // --- O += P @ V : 4 output d-tiles x (keys 0..31 / 32..63) ---
#pragma unroll
    for (int t2 = 0; t2 < 4; ++t2) {
#pragma unroll
      for (int ks = 0; ks < 2; ++ks) {
        const __bf16* bp = Vc + (t2 * 16 + l15) * 64 + ks * 32 + half * 8;
        V16 b;
        b.h[0] = *(const v8bf*)bp;
        b.h[1] = *(const v8bf*)(bp + 16);
        o[t2] = wmma_bf16(ks ? ap1.v : ap0.v, b.v, o[t2]);
      }
    }
    // top-of-loop wait+barrier protects buffer reuse; no trailing barrier
  }

  // --- normalize + store O (row n -> Ob[n*64 + d] == (2048 x 512) layout) ---
#pragma unroll
  for (int r = 0; r < 8; ++r) {
    const float inv = 1.0f / l[r];
    const size_t n = (size_t)(qbase + half * 8 + r) * DH;
    Ob[n +  0 + l15] = (__bf16)(o[0][r] * inv);
    Ob[n + 16 + l15] = (__bf16)(o[1][r] * inv);
    Ob[n + 32 + l15] = (__bf16)(o[2][r] * inv);
    Ob[n + 48 + l15] = (__bf16)(o[3][r] * inv);
  }
}

// ---------------------------------------------------------------------------
// 4) Final projection: f32 out = A[2048x512]bf16 @ WoutT^T + bias
// ---------------------------------------------------------------------------
__global__ __launch_bounds__(128) void k_gemm_out(
    const __bf16* __restrict__ A, const __bf16* __restrict__ BT,
    const float* __restrict__ bias, float* __restrict__ out,
    int M, int N, int K) {
  const int lane = threadIdx.x & 31;
  const int wv   = threadIdx.x >> 5;
  const int w    = blockIdx.x * 4 + wv;
  const int strips_n = N >> 6;
  const int mt = w / strips_n;
  const int nt = w - mt * strips_n;
  if (mt >= (M >> 4)) return;

  const int half = lane >> 4, l15 = lane & 15;
  const int kofs = half * 8;
  const size_t arow = (size_t)(mt * 16 + l15) * K;

  v8f c[4] = {v8f{}, v8f{}, v8f{}, v8f{}};
  for (int kk = 0; kk < K; kk += 32) {
    V16 a;
    a.h[0] = *(const v8bf*)(A + arow + kk + kofs);
    a.h[1] = *(const v8bf*)(A + arow + kk + 16 + kofs);
    const int koff = kk + half * 8;
#pragma unroll
    for (int t = 0; t < 4; ++t) {
      const __bf16* bp = BT + (size_t)(nt * 64 + t * 16 + l15) * K + koff;
      V16 b;
      b.h[0] = *(const v8bf*)bp;
      b.h[1] = *(const v8bf*)(bp + 16);
      c[t] = wmma_bf16(a.v, b.v, c[t]);
    }
  }

#pragma unroll
  for (int t = 0; t < 4; ++t) {
    const int col = nt * 64 + t * 16 + l15;
    const float bv = bias[col];
#pragma unroll
    for (int r = 0; r < 8; ++r) {
      const int row = mt * 16 + half * 8 + r;
      out[(size_t)row * N + col] = c[t][r] + bv;
    }
  }
}

// ---------------------------------------------------------------------------
// Host-side launch
// ---------------------------------------------------------------------------
extern "C" void kernel_launch(void* const* d_in, const int* in_sizes, int n_in,
                              void* d_out, int out_size, void* d_ws,
                              size_t ws_size, hipStream_t stream) {
  const float* x    = (const float*)d_in[0];   // 2048 x 1024
  const float* Wq   = (const float*)d_in[1];   // 1024 x 512
  const float* Wkv  = (const float*)d_in[2];   // 1024 x 1024
  const float* Wout = (const float*)d_in[3];   // 512 x 1024
  const float* bout = (const float*)d_in[4];   // 1024
  float* out = (float*)d_out;                  // 2048 x 1024

  // Workspace carve (bf16 buffers), 16 MiB total
  char* ws = (char*)d_ws;
  __bf16* xb    = (__bf16*)(ws);                         // 2048x1024   (4 MiB)
  __bf16* wqT   = (__bf16*)(ws + (4u << 20));            // 512x1024    (1 MiB)
  __bf16* wkvT  = (__bf16*)(ws + (5u << 20));            // 1024x1024   (2 MiB)
  __bf16* woutT = (__bf16*)(ws + (7u << 20));            // 1024x512    (1 MiB)
  __bf16* Qb    = (__bf16*)(ws + (8u << 20));            // 16384x64    (2 MiB)
  __bf16* Kb    = (__bf16*)(ws + (10u << 20));           // 16384x64    (2 MiB)
  __bf16* Vtb   = (__bf16*)(ws + (12u << 20));           // 64x16384    (2 MiB)
  __bf16* Ob    = (__bf16*)(ws + (14u << 20));           // 2048x512    (2 MiB)

  // 1) convert / transpose weights + activations to bf16
  k_cvt<<<2048, 256, 0, stream>>>(x, xb, BATCH * DMODEL);
  k_cvt_tr<<<1024, 256, 0, stream>>>(Wq, wqT, DMODEL, INNER);
  k_cvt_tr<<<2048, 256, 0, stream>>>(Wkv, wkvT, DMODEL, 2 * INNER);
  k_cvt_tr<<<1024, 256, 0, stream>>>(Wout, woutT, INNER, DMODEL);

  // 2) projections: Q (scale folded), K, V (written transposed)
  const float qscale = 0.125f;  // DH^-0.5 = 64^-0.5
  k_gemm_bf16<<<256, 128, 0, stream>>>(xb, wqT, Qb, BATCH, INNER, DMODEL,
                                       INNER, qscale, 0);
  k_gemm_bf16<<<256, 128, 0, stream>>>(xb, wkvT, Kb, BATCH, INNER, DMODEL,
                                       INNER, 1.0f, 0);
  k_gemm_bf16<<<256, 128, 0, stream>>>(xb, wkvT + (size_t)INNER * DMODEL, Vtb,
                                       BATCH, INNER, DMODEL, 0, 1.0f, 1);

  // 3) flash attention over N=16384, d=64
  k_flash<<<NQ / 64, 128, 0, stream>>>(Qb, Kb, Vtb, Ob);

  // 4) output projection + bias -> fp32 d_out
  k_gemm_out<<<512, 128, 0, stream>>>(Ob, woutT, bout, out, BATCH, DMODEL,
                                      INNER);
}
